// ProposalModule_21784074125442
// MI455X (gfx1250) — matrix operand order repaired and checked by reading, not compiled, not run on hardware
//
#include <hip/hip_runtime.h>

typedef __attribute__((ext_vector_type(16))) _Float16 v16h;
typedef __attribute__((ext_vector_type(8)))  float    v8f;

#define PI_F 3.14159265358979323846f

constexpr int B_  = 512;
constexpr int P_  = 256;
constexpr int BP_ = B_ * P_;
constexpr float EPSV = 1e-5f;

// LDS strides. W rows padded so uint index = m*(odd) + q -> bank (m+q)%64,
// conflict-free across the 16 rows a fragment load touches.
constexpr int WS1_STRIDE_U = 129;  // gemm1: 256 K halves + 2 pad (uints per row)
constexpr int WS2_STRIDE_U = 65;   // gemm2/3: 128 K halves + 2 pad
constexpr int XS_STRIDE   = 40;    // X transposed [n][k]: 32 halves + 8 pad

union AFrag { unsigned u[8]; v16h v; };
union H2    { unsigned u; _Float16 h[2]; };

// ---------------------------------------------------------------------------
// CDNA5 async global->LDS copy (ASYNCcnt-tracked), ISA §10 / §15.18.3.
// Low 32 bits of a generic shared pointer are the wave LDS byte offset.
// ---------------------------------------------------------------------------
__device__ inline void async_ld_b128(void* lds_ptr, const void* gaddr) {
  asm volatile("global_load_async_to_lds_b128 %0, %1, off"
               :: "v"((unsigned)(size_t)lds_ptr),
                  "v"((unsigned long long)(size_t)gaddr)
               : "memory");
}
__device__ inline void wait_async0() {
  asm volatile("s_wait_asynccnt 0x0" ::: "memory");
}

// A-matrix 16x32 (MxK), ISA 7.12.2. WsuRow points at the k0 pair of row m.
__device__ inline v16h load_a_frag(const unsigned* WsuRow, int kh) {
  AFrag af;
#pragma unroll
  for (int j = 0; j < 8; ++j)
    af.u[j] = WsuRow[(j & 3) + ((j >> 2) << 3) + (kh << 2)];
  return af.v;
}

// B-matrix 32x16 (KxN): element e holds K = kh*8+e (e<8), 16+kh*8+(e-8) (e>=8), col = lane&15.
// X transposed [n][k] (20 uints/row) -> K pairs are adjacent uints.
__device__ inline v16h load_b_frag(const unsigned* Xsu, int n, int kh) {
  AFrag bf;
#pragma unroll
  for (int j = 0; j < 8; ++j)
    bf.u[j] = Xsu[n * (XS_STRIDE / 2) + (j & 3) + ((j >> 2) << 3) + (kh << 2)];
  return bf.v;
}

// ---------------------------------------------------------------------------
// prep: weights f32->f16 (W3 zero-padded 97->112 rows), zero BN accumulators
// ---------------------------------------------------------------------------
__global__ __launch_bounds__(256) void prep_kernel(
    const float* __restrict__ W1, const float* __restrict__ W2,
    const float* __restrict__ W3,
    _Float16* __restrict__ W1h, _Float16* __restrict__ W2h,
    _Float16* __restrict__ W3h, float* __restrict__ accum) {
  int id = blockIdx.x * 256 + threadIdx.x;
  if (id < 32768) {
    W1h[id] = (_Float16)W1[id];
  } else if (id < 49152) {
    int j = id - 32768;
    W2h[j] = (_Float16)W2[j];
  } else if (id < 63488) {
    int j = id - 49152;
    int m = j >> 7, kk = j & 127;
    W3h[j] = (m < 97) ? (_Float16)W3[m * 128 + kk] : (_Float16)0.f;
  } else if (id < 64000) {
    accum[id - 63488] = 0.f;
  }
}

// ---------------------------------------------------------------------------
// GEMM1: y1h(f16) = W1(128x256) @ concat(features,rn) + b1, + BN1 stats.
// 8 waves, M=128, N=64 (4 subtiles/wave), K-step 32, 8 steps.
// Full W in LDS; async X copies double-buffered and software-pipelined.
// ---------------------------------------------------------------------------
__global__ __launch_bounds__(256) void gemm1_kernel(
    const float* __restrict__ features, const float* __restrict__ rn,
    const _Float16* __restrict__ Wh, const float* __restrict__ bias,
    _Float16* __restrict__ y, float* __restrict__ gsum, float* __restrict__ gsq) {
  __shared__ _Float16 Ws[128 * 2 * WS1_STRIDE_U];   // full 128x256 + pad
  __shared__ _Float16 Xs[64 * XS_STRIDE];
  __shared__ float    Xf[2][32 * 64];               // raw f32 tiles (pipelined)
  __shared__ float    s_sum[128], s_sq[128];
  const int t = threadIdx.x, wave = t >> 5, lane = t & 31;
  const int b = blockIdx.y, p0 = blockIdx.x * 64;
  const int kh = lane >> 4, nl = lane & 15;
  if (t < 128) { s_sum[t] = 0.f; s_sq[t] = 0.f; }
  v8f acc[4] = {{}, {}, {}, {}};
  unsigned* Wsu = (unsigned*)Ws;
  unsigned* Xsu = (unsigned*)Xs;
  const unsigned* Wg = (const unsigned*)Wh;         // 128 uints per row

  // issue async copy of K-step 0 (overlaps with W staging below)
#pragma unroll
  for (int h = 0; h < 2; ++h) {
    int i0 = t * 8 + h * 4;
    int k = i0 >> 6, n = i0 & 63;
    const float* src = (k < 128)
        ? (features + (size_t)(b * 128 + k) * P_ + p0 + n)
        : (rn + (size_t)(b * 128 + (k - 128)) * P_ + p0 + n);   // k<32 here, but keep form
    async_ld_b128(&Xf[0][i0], src);
  }
  // stage full W once: 128 rows x 128 uints
#pragma unroll 4
  for (int i = t; i < 16384; i += 256) {
    int m = i >> 7, q = i & 127;
    Wsu[m * WS1_STRIDE_U + q] = Wg[m * 128 + q];
  }

  for (int ks = 0; ks < 8; ++ks) {
    const int k0 = ks * 32;
    const int buf = ks & 1;
    wait_async0();                    // K-step ks landed (next not yet issued)
    __syncthreads();                  // all waves' async data + W visible
    if (ks + 1 < 8) {                 // pipeline: issue ks+1 during convert+compute
      const int k0n = k0 + 32;
#pragma unroll
      for (int h = 0; h < 2; ++h) {
        int i0 = t * 8 + h * 4;
        int k = (i0 >> 6) + k0n, n = i0 & 63;
        const float* src = (k < 128)
            ? (features + (size_t)(b * 128 + k) * P_ + p0 + n)
            : (rn + (size_t)(b * 128 + (k - 128)) * P_ + p0 + n);
        async_ld_b128(&Xf[buf ^ 1][i0], src);
      }
    }
    // convert f32 -> f16, transpose into [n][k]
#pragma unroll
    for (int i = t; i < 2048; i += 256) {
      int k = i >> 6, n = i & 63;
      Xs[n * XS_STRIDE + k] = (_Float16)Xf[buf][i];
    }
    __syncthreads();
    v16h a = load_a_frag(Wsu + (wave * 16 + nl) * WS1_STRIDE_U + (k0 >> 1), kh);
#pragma unroll
    for (int s = 0; s < 4; ++s) {
      v16h bb = load_b_frag(Xsu, s * 16 + nl, kh);
      acc[s] = __builtin_amdgcn_wmma_f32_16x16x32_f16(false, a, false, bb,
                                                      (short)0, acc[s], false, false);
    }
  }
  // epilogue: bias, f16 store, channel stats
#pragma unroll
  for (int r = 0; r < 8; ++r) {
    int m = wave * 16 + r + (kh << 3);
    float bm = bias[m];
    size_t rowbase = (size_t)(b * 128 + m) * P_ + p0 + nl;
    float s = 0.f, q = 0.f;
#pragma unroll
    for (int ss = 0; ss < 4; ++ss) {
      float v = acc[ss][r] + bm;
      y[rowbase + ss * 16] = (_Float16)v;
      s += v; q += v * v;
    }
#pragma unroll
    for (int msk = 1; msk < 16; msk <<= 1) {
      s += __shfl_xor(s, msk, 32);
      q += __shfl_xor(q, msk, 32);
    }
    if (nl == 0) { atomicAdd(&s_sum[m], s); atomicAdd(&s_sq[m], q); }
  }
  __syncthreads();
  if (t < 128) { atomicAdd(&gsum[t], s_sum[t]); atomicAdd(&gsq[t], s_sq[t]); }
}

// ---------------------------------------------------------------------------
__global__ void bnparam_kernel(const float* __restrict__ gsum,
                               const float* __restrict__ gsq,
                               const float* __restrict__ gamma,
                               const float* __restrict__ beta,
                               float* __restrict__ scale,
                               float* __restrict__ shift) {
  int t = threadIdx.x;
  if (t < 128) {
    float invN = 1.0f / (float)BP_;
    float m = gsum[t] * invN;
    float var = gsq[t] * invN - m * m;
    float sc = gamma[t] * rsqrtf(var + EPSV);
    scale[t] = sc;
    shift[t] = beta[t] - m * sc;
  }
}

// ---------------------------------------------------------------------------
// GEMM2: y2h = W2(128x128) @ relu(bn(y1h)) + b2, + BN2 stats.
// ---------------------------------------------------------------------------
__global__ __launch_bounds__(256) void gemm2_kernel(
    const _Float16* __restrict__ x, const float* __restrict__ scale,
    const float* __restrict__ shift,
    const _Float16* __restrict__ Wh, const float* __restrict__ bias,
    _Float16* __restrict__ y, float* __restrict__ gsum, float* __restrict__ gsq) {
  __shared__ _Float16 Ws[128 * 2 * WS2_STRIDE_U];   // full 128x128 + pad
  __shared__ _Float16 Xs[64 * XS_STRIDE];
  __shared__ unsigned Xr[2][32 * 32];               // raw f16 tiles (pipelined)
  __shared__ float    s_sum[128], s_sq[128];
  const int t = threadIdx.x, wave = t >> 5, lane = t & 31;
  const int b = blockIdx.y, p0 = blockIdx.x * 64;
  const int kh = lane >> 4, nl = lane & 15;
  if (t < 128) { s_sum[t] = 0.f; s_sq[t] = 0.f; }
  v8f acc[4] = {{}, {}, {}, {}};
  unsigned* Wsu = (unsigned*)Ws;
  unsigned* Xsu = (unsigned*)Xs;
  const unsigned* Wg = (const unsigned*)Wh;         // 64 uints per row
  const unsigned* Xg = (const unsigned*)x;

  {                                                 // issue K-step 0
    int i0 = t * 4;
    int k = i0 >> 5, j = i0 & 31;
    async_ld_b128(&Xr[0][i0],
                  Xg + (((size_t)(b * 128 + k) * P_ + p0) >> 1) + j);
  }
#pragma unroll 4
  for (int i = t; i < 8192; i += 256) {             // full W once
    int m = i >> 6, q = i & 63;
    Wsu[m * WS2_STRIDE_U + q] = Wg[m * 64 + q];
  }

  for (int ks = 0; ks < 4; ++ks) {
    const int k0 = ks * 32;
    const int buf = ks & 1;
    wait_async0();
    __syncthreads();
    if (ks + 1 < 4) {
      int i0 = t * 4;
      int k = (i0 >> 5) + k0 + 32, j = i0 & 31;
      async_ld_b128(&Xr[buf ^ 1][i0],
                    Xg + (((size_t)(b * 128 + k) * P_ + p0) >> 1) + j);
    }
#pragma unroll
    for (int i = t; i < 1024; i += 256) {           // BN+ReLU, transpose [n][k]
      int k = i >> 5, j = i & 31, cch = k0 + k, n = j * 2;
      H2 u; u.u = Xr[buf][i];
      float sc = scale[cch], sh = shift[cch];
      float v0 = fmaxf((float)u.h[0] * sc + sh, 0.f);
      float v1 = fmaxf((float)u.h[1] * sc + sh, 0.f);
      Xs[n * XS_STRIDE + k]       = (_Float16)v0;
      Xs[(n + 1) * XS_STRIDE + k] = (_Float16)v1;
    }
    __syncthreads();
    v16h a = load_a_frag(Wsu + (wave * 16 + nl) * WS2_STRIDE_U + (k0 >> 1), kh);
#pragma unroll
    for (int s = 0; s < 4; ++s) {
      v16h bb = load_b_frag(Xsu, s * 16 + nl, kh);
      acc[s] = __builtin_amdgcn_wmma_f32_16x16x32_f16(false, a, false, bb,
                                                      (short)0, acc[s], false, false);
    }
  }
#pragma unroll
  for (int r = 0; r < 8; ++r) {
    int m = wave * 16 + r + (kh << 3);
    float bm = bias[m];
    size_t rowbase = (size_t)(b * 128 + m) * P_ + p0 + nl;
    float s = 0.f, q = 0.f;
#pragma unroll
    for (int ss = 0; ss < 4; ++ss) {
      float v = acc[ss][r] + bm;
      y[rowbase + ss * 16] = (_Float16)v;
      s += v; q += v * v;
    }
#pragma unroll
    for (int msk = 1; msk < 16; msk <<= 1) {
      s += __shfl_xor(s, msk, 32);
      q += __shfl_xor(q, msk, 32);
    }
    if (nl == 0) { atomicAdd(&s_sum[m], s); atomicAdd(&s_sq[m], q); }
  }
  __syncthreads();
  if (t < 128) { atomicAdd(&gsum[t], s_sum[t]); atomicAdd(&gsq[t], s_sq[t]); }
}

// ---------------------------------------------------------------------------
// GEMM3: net = W3(97->112 x128) @ relu(bn(y2h)) + b3, decode fused + scatter.
// ---------------------------------------------------------------------------
__global__ __launch_bounds__(256) void gemm3_kernel(
    const _Float16* __restrict__ x, const float* __restrict__ scale,
    const float* __restrict__ shift,
    const _Float16* __restrict__ Wh /*112x128*/, const float* __restrict__ b3,
    const float* __restrict__ xyz, const float* __restrict__ msa,
    float* __restrict__ out) {
  __shared__ _Float16 Ws[112 * 2 * WS2_STRIDE_U];
  __shared__ _Float16 Xs[64 * XS_STRIDE];
  __shared__ unsigned Xr[2][32 * 32];
  const int t = threadIdx.x, wave = t >> 5, lane = t & 31;
  const int b = blockIdx.y, p0 = blockIdx.x * 64;
  const int kh = lane >> 4, nl = lane & 15;
  v8f acc[4] = {{}, {}, {}, {}};
  unsigned* Wsu = (unsigned*)Ws;
  unsigned* Xsu = (unsigned*)Xs;
  const unsigned* Wg = (const unsigned*)Wh;         // 64 uints per row
  const unsigned* Xg = (const unsigned*)x;

  {
    int i0 = t * 4;
    int k = i0 >> 5, j = i0 & 31;
    async_ld_b128(&Xr[0][i0],
                  Xg + (((size_t)(b * 128 + k) * P_ + p0) >> 1) + j);
  }
#pragma unroll 4
  for (int i = t; i < 7168; i += 256) {             // 112 rows x 64 uints
    int m = i >> 6, q = i & 63;
    Wsu[m * WS2_STRIDE_U + q] = Wg[m * 64 + q];
  }

  for (int ks = 0; ks < 4; ++ks) {
    const int k0 = ks * 32;
    const int buf = ks & 1;
    wait_async0();
    __syncthreads();
    if (ks + 1 < 4) {
      int i0 = t * 4;
      int k = (i0 >> 5) + k0 + 32, j = i0 & 31;
      async_ld_b128(&Xr[buf ^ 1][i0],
                    Xg + (((size_t)(b * 128 + k) * P_ + p0) >> 1) + j);
    }
#pragma unroll
    for (int i = t; i < 1024; i += 256) {
      int k = i >> 5, j = i & 31, cch = k0 + k, n = j * 2;
      H2 u; u.u = Xr[buf][i];
      float sc = scale[cch], sh = shift[cch];
      float v0 = fmaxf((float)u.h[0] * sc + sh, 0.f);
      float v1 = fmaxf((float)u.h[1] * sc + sh, 0.f);
      Xs[n * XS_STRIDE + k]       = (_Float16)v0;
      Xs[(n + 1) * XS_STRIDE + k] = (_Float16)v1;
    }
    __syncthreads();
    if (wave < 7) {                                 // waves 0..6 cover M=0..111
      v16h a = load_a_frag(Wsu + (wave * 16 + nl) * WS2_STRIDE_U + (k0 >> 1), kh);
#pragma unroll
      for (int s = 0; s < 4; ++s) {
        v16h bb = load_b_frag(Xsu, s * 16 + nl, kh);
        acc[s] = __builtin_amdgcn_wmma_f32_16x16x32_f16(false, a, false, bb,
                                                        (short)0, acc[s], false, false);
      }
    }
  }
  if (wave < 7) {
#pragma unroll
    for (int s = 0; s < 4; ++s) {
      const int p = p0 + s * 16 + nl;
      const size_t bp = (size_t)b * P_ + p;
#pragma unroll
      for (int r = 0; r < 8; ++r) {
        int o = wave * 16 + r + (kh << 3);
        if (o < 97) {
          float v = acc[s][r] + b3[o];
          size_t idx;
          if (o < 3)       { v += xyz[bp * 3 + o]; idx = bp * 3 + o; }
          else if (o == 3) { idx = (size_t)BP_ * 3 + bp; }
          else if (o == 4) { v *= PI_F; idx = (size_t)BP_ * 4 + bp; }
          else if (o < 23) { idx = (size_t)BP_ * 5 + bp * 18 + (o - 5); }
          else if (o < 77) { v *= msa[o - 23];
                             idx = (size_t)BP_ * 23 + bp * 54 + (o - 23); }
          else             { idx = (size_t)BP_ * 77 + bp * 20 + (o - 77); }
          out[idx] = v;
        }
      }
    }
  }
}

// ---------------------------------------------------------------------------
// idx_obj / sum_one: stable partition per batch row via inclusive scan
// ---------------------------------------------------------------------------
__global__ __launch_bounds__(256) void index_kernel(const int* __restrict__ label,
                                                    float* __restrict__ out) {
  __shared__ int sc[256];
  const int b = blockIdx.x, t = threadIdx.x;
  const int lab = label[b * P_ + t];
  sc[t] = lab;
  for (int off = 1; off < 256; off <<= 1) {
    __syncthreads();
    int v = (t >= off) ? sc[t - off] : 0;
    __syncthreads();
    sc[t] += v;
  }
  __syncthreads();
  const int total = sc[255];
  float* oidx = out + (size_t)BP_ * 97;
  if (lab) oidx[b * P_ + (sc[t] - 1)] = (float)t;
  if (t >= total) oidx[b * P_ + t] = (float)t;
  if (t == 0) out[(size_t)BP_ * 97 + BP_ + b] = (float)total;
}

// ---------------------------------------------------------------------------
extern "C" void kernel_launch(void* const* d_in, const int* in_sizes, int n_in,
                              void* d_out, int out_size, void* d_ws, size_t ws_size,
                              hipStream_t stream) {
  (void)in_sizes; (void)n_in; (void)out_size; (void)ws_size;
  const float* xyz      = (const float*)d_in[0];
  const float* features = (const float*)d_in[1];
  const float* rn       = (const float*)d_in[2];
  const int*   label    = (const int*)d_in[3];
  const float* W1  = (const float*)d_in[4];
  const float* b1  = (const float*)d_in[5];
  const float* g1  = (const float*)d_in[6];
  const float* be1 = (const float*)d_in[7];
  const float* W2  = (const float*)d_in[8];
  const float* b2  = (const float*)d_in[9];
  const float* g2  = (const float*)d_in[10];
  const float* be2 = (const float*)d_in[11];
  const float* W3  = (const float*)d_in[12];
  const float* b3  = (const float*)d_in[13];
  const float* msa = (const float*)d_in[14];

  char* ws = (char*)d_ws;                               // ~64.1 MiB used
  _Float16* y1h = (_Float16*)ws;                        // 32 MiB
  _Float16* y2h = (_Float16*)(ws + (size_t)33554432);   // 32 MiB
  _Float16* W1h = (_Float16*)(ws + (size_t)67108864);   // 64 KiB
  _Float16* W2h = (_Float16*)(ws + (size_t)67174400);   // 32 KiB
  _Float16* W3h = (_Float16*)(ws + (size_t)67207168);   // 28 KiB
  float*    acc = (float*)(ws + (size_t)67235840);      // 512 f (zeroed by prep)
  float*    bnp = (float*)(ws + (size_t)67237888);      // 512 f
  float *sum1 = acc,       *sq1 = acc + 128;
  float *sum2 = acc + 256, *sq2 = acc + 384;
  float *scale1 = bnp,       *shift1 = bnp + 128;
  float *scale2 = bnp + 256, *shift2 = bnp + 384;

  prep_kernel<<<250, 256, 0, stream>>>(W1, W2, W3, W1h, W2h, W3h, acc);

  dim3 grid(P_ / 64, B_);
  gemm1_kernel<<<grid, 256, 0, stream>>>(features, rn, W1h, b1, y1h, sum1, sq1);
  bnparam_kernel<<<1, 128, 0, stream>>>(sum1, sq1, g1, be1, scale1, shift1);
  gemm2_kernel<<<grid, 256, 0, stream>>>(y1h, scale1, shift1, W2h, b2, y2h, sum2, sq2);
  bnparam_kernel<<<1, 128, 0, stream>>>(sum2, sq2, g2, be2, scale2, shift2);
  gemm3_kernel<<<grid, 256, 0, stream>>>(y2h, scale2, shift2, W3h, b3, xyz, msa,
                                         (float*)d_out);
  index_kernel<<<B_, 256, 0, stream>>>(label, (float*)d_out);
}
